// LSTM_15144054686345
// MI455X (gfx1250) — matrix operand order repaired and checked by reading, not compiled, not run on hardware
//
#include <hip/hip_runtime.h>
#include <hip/hip_bf16.h>
#include <stdint.h>

// LSTM classifier: B=64, S=512, E=512, H=1024, NCLS=2
// Persistent-kernel bf16-WMMA LSTM for gfx1250 (MI455X):
//  - weights pre-converted to bf16, [Wx;Wh] concatenated, gate-interleaved cols
//  - 64 workgroups, each pins its 192KB weight slice in LDS (320KB/WGP)
//  - one grid barrier per timestep, double-buffered E/H bf16 staging in ws
//  - each wave: 2 col-tiles sharing one A-fragment stream, 2 independent
//    accumulators -> back-to-back independent v_wmma issue

#define B_   64
#define S_   512
#define E_   512
#define H_   1024
#define GC_  4096          // 4*H gate columns
#define K_   1536          // E + H
#define NWG  64
#define NTH  256

typedef __attribute__((ext_vector_type(16))) __bf16 v16bf;
typedef __attribute__((ext_vector_type(8)))  float  v8f;

union AF { v16bf v; uint4 q[2]; };

__device__ inline unsigned short f2bf(float f) {
    union { float f; unsigned u; } v; v.f = f;
    unsigned u = v.u;
    u += 0x7fffu + ((u >> 16) & 1u);      // round-to-nearest-even
    return (unsigned short)(u >> 16);
}

struct ushort4_ { unsigned short s0, s1, s2, s3; };

__device__ inline float sigm_(float v) { return 1.0f / (1.0f + __expf(-v)); }
__device__ inline float tanh_(float v) {
    float e = __expf(-2.0f * v);
    return (1.0f - e) / (1.0f + e);
}

__device__ inline void grid_barrier(unsigned* ctr) {
    __threadfence();              // flush this wave's global stores
    __syncthreads();
    if (threadIdx.x == 0) {
        unsigned old = atomicAdd(ctr, 1u);
        unsigned target = ((old / NWG) + 1u) * NWG;
        while (__hip_atomic_load(ctr, __ATOMIC_RELAXED, __HIP_MEMORY_SCOPE_AGENT) < target) {
            __builtin_amdgcn_s_sleep(4);
        }
    }
    __syncthreads();
    __threadfence();
}

// ---------------------------------------------------------------------------
// Prep: fp32 -> bf16 weight concat [gc=h*4+g][k in 0..1536), zero H buffers,
// reset the barrier counter (makes every kernel_launch call deterministic).
// ---------------------------------------------------------------------------
__global__ void prep_kernel(const float* __restrict__ Wii, const float* __restrict__ Whi,
                            const float* __restrict__ Wif, const float* __restrict__ Whf,
                            const float* __restrict__ Wig, const float* __restrict__ Whg,
                            const float* __restrict__ Wio, const float* __restrict__ Who,
                            unsigned short* __restrict__ Wcat,
                            unsigned short* __restrict__ Hbuf,
                            unsigned* __restrict__ ctr)
{
    int gtid = blockIdx.x * blockDim.x + threadIdx.x;
    int nthr = gridDim.x * blockDim.x;
    for (int idx = gtid; idx < GC_ * K_; idx += nthr) {
        int gc = idx / K_;
        int k  = idx - gc * K_;
        int h  = gc >> 2, g = gc & 3;
        const float* Wx = (g == 0) ? Wii : (g == 1) ? Wif : (g == 2) ? Wig : Wio;
        const float* Wh = (g == 0) ? Whi : (g == 1) ? Whf : (g == 2) ? Whg : Who;
        float v = (k < E_) ? Wx[(size_t)h * E_ + k] : Wh[(size_t)h * H_ + (k - E_)];
        Wcat[idx] = f2bf(v);
    }
    for (int idx = gtid; idx < 2 * B_ * H_; idx += nthr) Hbuf[idx] = 0;
    if (gtid == 0) *ctr = 0u;
}

// ---------------------------------------------------------------------------
// Persistent LSTM kernel: 64 WGs x 256 threads (8 wave32s).
// WG wg owns gate columns [wg*64, wg*64+64) == h-cols [wg*16, wg*16+16) x 4 gates.
// ---------------------------------------------------------------------------
__global__ void __launch_bounds__(NTH, 1)
lstm_persistent(const int* __restrict__ x, const float* __restrict__ emb,
                const float* __restrict__ b_i, const float* __restrict__ b_f,
                const float* __restrict__ b_g, const float* __restrict__ b_o,
                const float* __restrict__ Vw,  const float* __restrict__ Vb,
                const unsigned short* __restrict__ Wcat,
                unsigned short* __restrict__ Ebuf,   // [2][B_*E_] bf16
                unsigned short* __restrict__ Hbuf,   // [2][B_*H_] bf16
                unsigned* __restrict__ ctr,
                float* __restrict__ out)
{
    extern __shared__ char smem[];
    unsigned short* lds_w = (unsigned short*)smem;                    // 64 x K_ bf16 = 192KB
    float* gbuf  = (float*)(smem + 64 * K_ * 2);                      // 64 x 64 gate pre-acts
    float* c_lds = (float*)(smem + 64 * K_ * 2 + 64 * 64 * 4);        // 64 x 16 cell state

    const int tid = threadIdx.x;
    const int wg  = blockIdx.x;       // 0..63
    const int gc0 = wg * 64;
    const int h0  = wg * 16;

    // Stage this WG's weight slice into LDS once; reused for all 512 steps.
    {
        const uint4* src = (const uint4*)(Wcat + (size_t)gc0 * K_);
        uint4* dst = (uint4*)lds_w;
        for (int i = tid; i < (64 * K_ * 2) / 16; i += NTH) dst[i] = src[i];
    }
    for (int i = tid; i < 64 * 16; i += NTH) c_lds[i] = 0.0f;
    __syncthreads();

    const int w    = tid >> 5;        // wave id 0..7
    const int l    = tid & 31;        // lane
    const int half = l >> 4;          // lane-half (contiguous K split: k = 16*half + s)
    const int ln   = l & 15;

    // Wave -> tile map: both tiles share the A row-tile (one A stream, 2 accs)
    const int rt  = w >> 1;           // row tile 0..3
    const int ct0 = (w & 1) * 2;      // col tiles ct0, ct0+1
    const int m   = rt * 16 + ln;     // batch row for A lanes
    const uint4* Bb0 = (const uint4*)(lds_w + (size_t)(ct0 * 16 + ln) * K_);
    const uint4* Bb1 = (const uint4*)(lds_w + (size_t)((ct0 + 1) * 16 + ln) * K_);

    float* out_logit = out;
    float* out_h     = out + B_ * 2;
    float* out_c     = out_h + B_ * H_;

    for (int t = 0; t < S_; ++t) {
        const int wr = t & 1;
        const int rd = wr ^ 1;
        unsigned short*       Ew = Ebuf + (size_t)wr * (B_ * E_);
        const unsigned short* Hr = Hbuf + (size_t)rd * (B_ * H_);
        unsigned short*       Hw = Hbuf + (size_t)wr * (B_ * H_);

        // 1) gather+convert embeddings for timestep t (vectorized x4, all WGs)
        {
            int gtid = wg * NTH + tid;                       // 16384 threads
            for (int e4 = gtid; e4 < (B_ * E_) / 4; e4 += NWG * NTH) {
                int e   = e4 * 4;
                int b   = e >> 9;                            // / E_
                int col = e & (E_ - 1);
                int tok = x[b * S_ + t];
                float4 v = *(const float4*)(emb + (size_t)tok * E_ + col);
                ushort4_ o;
                o.s0 = f2bf(v.x); o.s1 = f2bf(v.y);
                o.s2 = f2bf(v.z); o.s3 = f2bf(v.w);
                *(ushort4_*)(Ew + e) = o;
            }
        }
        grid_barrier(ctr);   // e_t staged, h_{t-1} staged, WAR on buffers cleared

        // 2) GEMM [64 x 64] = [64 x 1536]bf16 . Wslice
        v8f acc0 = {}, acc1 = {};

        // E region (k = 0..512)
        const uint4* Ab = (const uint4*)(Ew + (size_t)m * E_);
        #pragma unroll 4
        for (int k0 = 0; k0 < E_; k0 += 32) {
            AF a, bm0, bm1;
            a.q[0]   = Ab[(k0 + half * 16) >> 3];
            a.q[1]   = Ab[(k0 + half * 16 + 8) >> 3];
            bm0.q[0] = Bb0[(k0 + half * 16) >> 3];
            bm0.q[1] = Bb0[(k0 + half * 16 + 8) >> 3];
            bm1.q[0] = Bb1[(k0 + half * 16) >> 3];
            bm1.q[1] = Bb1[(k0 + half * 16 + 8) >> 3];
            acc0 = __builtin_amdgcn_wmma_f32_16x16x32_bf16(
                       false, a.v, false, bm0.v, (short)0, acc0, false, false);
            acc1 = __builtin_amdgcn_wmma_f32_16x16x32_bf16(
                       false, a.v, false, bm1.v, (short)0, acc1, false, false);
        }
        // H region (k = 512..1536)
        const uint4* Ah = (const uint4*)(Hr + (size_t)m * H_);
        #pragma unroll 4
        for (int k0 = 0; k0 < H_; k0 += 32) {
            AF a, bm0, bm1;
            int kc   = E_ + k0;
            a.q[0]   = Ah[(k0 + half * 16) >> 3];
            a.q[1]   = Ah[(k0 + half * 16 + 8) >> 3];
            bm0.q[0] = Bb0[(kc + half * 16) >> 3];
            bm0.q[1] = Bb0[(kc + half * 16 + 8) >> 3];
            bm1.q[0] = Bb1[(kc + half * 16) >> 3];
            bm1.q[1] = Bb1[(kc + half * 16 + 8) >> 3];
            acc0 = __builtin_amdgcn_wmma_f32_16x16x32_bf16(
                       false, a.v, false, bm0.v, (short)0, acc0, false, false);
            acc1 = __builtin_amdgcn_wmma_f32_16x16x32_bf16(
                       false, a.v, false, bm1.v, (short)0, acc1, false, false);
        }
        // C/D layout: VGPR r holds (m_local = r + half*8, n = ln)
        #pragma unroll
        for (int r = 0; r < 8; ++r) {
            int row = rt * 16 + r + half * 8;
            gbuf[row * 64 + ct0 * 16 + ln]       = acc0[r];
            gbuf[row * 64 + (ct0 + 1) * 16 + ln] = acc1[r];
        }
        __syncthreads();

        // 3) gates -> activations -> state update (cols interleaved h*4+g)
        for (int e = tid; e < B_ * 16; e += NTH) {
            int b  = e >> 4;
            int hl = e & 15;
            int hcol = h0 + hl;
            float gi = gbuf[b * 64 + hl * 4 + 0] + b_i[hcol];
            float gf = gbuf[b * 64 + hl * 4 + 1] + b_f[hcol];
            float gg = gbuf[b * 64 + hl * 4 + 2] + b_g[hcol];
            float go = gbuf[b * 64 + hl * 4 + 3] + b_o[hcol];
            float it = sigm_(gi), ft = sigm_(gf), ot = sigm_(go);
            float gt = tanh_(gg);
            float c  = ft * c_lds[e] + it * gt;
            float h  = ot * tanh_(c);
            c_lds[e] = c;
            Hw[b * H_ + hcol] = f2bf(h);
            if (t == S_ - 1) {
                out_h[b * H_ + hcol] = h;
                out_c[b * H_ + hcol] = c;
            }
        }
        // next iteration's grid_barrier provides the needed block+grid sync
    }

    grid_barrier(ctr);   // all final h visible
    if (wg == 0 && tid < B_ * 2) {
        int b = tid >> 1, cls = tid & 1;
        float acc = Vb[cls];
        for (int k2 = 0; k2 < H_; ++k2)
            acc += out_h[b * H_ + k2] * Vw[cls * H_ + k2];
        out_logit[b * 2 + cls] = acc;
    }
}

// ---------------------------------------------------------------------------
extern "C" void kernel_launch(void* const* d_in, const int* in_sizes, int n_in,
                              void* d_out, int out_size, void* d_ws, size_t ws_size,
                              hipStream_t stream) {
    const int*   x   = (const int*)  d_in[0];
    const float* emb = (const float*)d_in[1];
    const float* Wii = (const float*)d_in[2];
    const float* bii = (const float*)d_in[3];
    const float* Whi = (const float*)d_in[4];
    const float* Wif = (const float*)d_in[5];
    const float* bif = (const float*)d_in[6];
    const float* Whf = (const float*)d_in[7];
    const float* Wig = (const float*)d_in[8];
    const float* big = (const float*)d_in[9];
    const float* Whg = (const float*)d_in[10];
    const float* Wio = (const float*)d_in[11];
    const float* bio = (const float*)d_in[12];
    const float* Who = (const float*)d_in[13];
    const float* Vw  = (const float*)d_in[14];
    const float* Vb  = (const float*)d_in[15];

    char* ws = (char*)d_ws;
    unsigned short* Wcat = (unsigned short*)ws;                               // 12,582,912 B
    unsigned short* Ebuf = (unsigned short*)(ws + 12582912);                  //    131,072 B (2 bufs)
    unsigned short* Hbuf = (unsigned short*)(ws + 12582912 + 131072);         //    262,144 B (2 bufs)
    unsigned*       ctr  = (unsigned*)      (ws + 12582912 + 131072 + 262144);

    prep_kernel<<<256, 256, 0, stream>>>(Wii, Whi, Wif, Whf, Wig, Whg, Wio, Who,
                                         Wcat, Hbuf, ctr);

    size_t shmem = (size_t)64 * K_ * 2 + 64 * 64 * 4 + 64 * 16 * 4;  // 217,088 B < 320KB WGP LDS
    hipFuncSetAttribute((const void*)lstm_persistent,
                        hipFuncAttributeMaxDynamicSharedMemorySize, (int)shmem);
    lstm_persistent<<<NWG, NTH, shmem, stream>>>(x, emb, bii, bif, big, bio, Vw, Vb,
                                                 Wcat, Ebuf, Hbuf, ctr, (float*)d_out);
}
